// TransformerBlock_40192303956213
// MI455X (gfx1250) — matrix operand order repaired and checked by reading, not compiled, not run on hardware
//
#include <hip/hip_runtime.h>

// ---------------------------------------------------------------------------
// Problem constants: B=2, T=2048, D=1024, H=16, HD=64, DFF=2730 (pad->2816)
// ---------------------------------------------------------------------------
#define TT    2048
#define DD    1024
#define NH    16
#define HDIM  64
#define D3    3072
#define DFFN  2730
#define DFFP  2816            // padded to 22*128 so every GEMM dim is tile-aligned
#define BT    (2 * TT)        // 4096 rows (B*T)

typedef __bf16 bf16;
typedef bf16  v8bf  __attribute__((ext_vector_type(8)));
typedef bf16  v16bf __attribute__((ext_vector_type(16)));
typedef float v8f   __attribute__((ext_vector_type(8)));

static __device__ __forceinline__ v8f zero8() {
    v8f z = {0.f, 0.f, 0.f, 0.f, 0.f, 0.f, 0.f, 0.f};
    return z;
}
static __device__ __forceinline__ v16bf cat8(v8bf lo, v8bf hi) {
    return __builtin_shufflevector(lo, hi, 0, 1, 2, 3, 4, 5, 6, 7,
                                           8, 9, 10, 11, 12, 13, 14, 15);
}
static __device__ __forceinline__ v8bf ld8(const bf16* p) {
    return *(const v8bf*)p;   // caller guarantees 16B alignment
}
static __device__ __forceinline__ v8f wmma_bf16(v16bf a, v16bf b, v8f c) {
    return __builtin_amdgcn_wmma_f32_16x16x32_bf16(
        false, a, false, b, (short)0, c, false, false);
}

// Async DMA of 64 contiguous bytes (per lane) global -> LDS, ASYNCcnt-tracked.
// INST_OFFSET is added to BOTH the LDS and global addresses (ISA 8 §4.4), so a
// single (lds_off, gaddr) pair covers all four 16B beats.
static __device__ __forceinline__ void async_copy64(unsigned lds_off,
                                                    const void* gptr) {
    asm volatile(
        "global_load_async_to_lds_b128 %0, %1, off\n\t"
        "global_load_async_to_lds_b128 %0, %1, off offset:16\n\t"
        "global_load_async_to_lds_b128 %0, %1, off offset:32\n\t"
        "global_load_async_to_lds_b128 %0, %1, off offset:48"
        :: "v"(lds_off), "v"((unsigned long long)(size_t)gptr)
        : "memory");
}
static __device__ __forceinline__ void wait_async0() {
    asm volatile("s_wait_asynccnt 0x0" ::: "memory");
}

// ---------------------------------------------------------------------------
// Tiled transpose + fp32->bf16 + zero-pad:  W[K,N] (f32) -> Wt[Npad,Kpad] (bf16)
// ---------------------------------------------------------------------------
__global__ __launch_bounds__(256) void transpose_pad_bf16_kernel(
    const float* __restrict__ W, int K, int N, bf16* __restrict__ Wt, int Kpad) {
    __shared__ float tile[32][33];
    const int k0 = blockIdx.x * 32;
    const int n0 = blockIdx.y * 32;
    const int tx = threadIdx.x & 31;
    const int ty = threadIdx.x >> 5;   // 0..7
    #pragma unroll
    for (int j = 0; j < 4; ++j) {
        const int k = k0 + ty + j * 8;
        const int n = n0 + tx;
        tile[ty + j * 8][tx] = (k < K && n < N) ? W[(size_t)k * N + n] : 0.f;
    }
    __syncthreads();
    #pragma unroll
    for (int j = 0; j < 4; ++j) {
        const int n = n0 + ty + j * 8;
        const int k = k0 + tx;
        Wt[(size_t)n * Kpad + k] = (bf16)tile[tx][ty + j * 8];
    }
}

// ---------------------------------------------------------------------------
// LayerNorm over last dim (D=1024), one row per block, 256 threads (8 waves)
// ---------------------------------------------------------------------------
__global__ __launch_bounds__(256) void layernorm_kernel(
    const float* __restrict__ x, const float* __restrict__ g,
    const float* __restrict__ b, bf16* __restrict__ out) {
    const int row  = blockIdx.x;
    const int tid  = threadIdx.x;
    const int lane = tid & 31;
    const int w    = tid >> 5;
    const size_t base = (size_t)row * DD;

    float s = 0.f, s2 = 0.f;
    #pragma unroll
    for (int j = 0; j < 4; ++j) {
        float v = x[base + tid + j * 256];
        s += v;
        s2 += v * v;
    }
    #pragma unroll
    for (int off = 16; off >= 1; off >>= 1) {
        s  += __shfl_xor(s, off);
        s2 += __shfl_xor(s2, off);
    }
    __shared__ float red0[8], red1[8];
    if (lane == 0) { red0[w] = s; red1[w] = s2; }
    __syncthreads();
    if (tid == 0) {
        float ts = 0.f, ts2 = 0.f;
        #pragma unroll
        for (int i = 0; i < 8; ++i) { ts += red0[i]; ts2 += red1[i]; }
        float mean = ts / (float)DD;
        float var  = ts2 / (float)DD - mean * mean;
        red0[0] = mean;
        red1[0] = rsqrtf(var + 1e-5f);
    }
    __syncthreads();
    const float mean = red0[0];
    const float rinv = red1[0];
    #pragma unroll
    for (int j = 0; j < 4; ++j) {
        const int i = tid + j * 256;
        out[base + i] = (bf16)((x[base + i] - mean) * rinv * g[i] + b[i]);
    }
}

// ---------------------------------------------------------------------------
// WMMA GEMM:  C[M,N] = A[M,K] @ Wt[N,K]^T      (all bf16, fully tile-aligned)
// Double-buffered LDS filled by ASYNC global->LDS DMA; WMMA on the live buffer
// overlaps the DMA of the next K-tile.  Block tile 128x128, 256 thr = 8 waves
// (4x2), wave tile 32x64, BK=64. All dims padded -> no guards anywhere.
//   mode 0: Cbf[row*ldc+col] = C (bf16)
//   mode 1: Cf [row*ldc+col] = resid[..] + C (fp32 residual add)
// ---------------------------------------------------------------------------
#define BM 128
#define BN 128
#define BK 64
#define LP 72     // LDS pitch in halves (144B, 16B-aligned, conflict-skewed)

__global__ __launch_bounds__(256) void gemm_bf16_kernel(
    const bf16* __restrict__ A, int lda,
    const bf16* __restrict__ Wt, int ldw,
    int K,
    bf16* __restrict__ Cbf, const float* __restrict__ resid,
    float* __restrict__ Cf, int ldc, int mode) {

    __shared__ bf16 As[2][BM * LP];    // [buf][row][k]
    __shared__ bf16 Bts[2][BM * LP];   // [buf][col][k]

    const int tid  = threadIdx.x;
    const int lane = tid & 31;
    const int w    = tid >> 5;
    const int m0   = blockIdx.y * BM;
    const int n0   = blockIdx.x * BN;
    const int wm   = (w >> 1) * 32;
    const int wn   = (w & 1) * 64;

    const int srow = tid >> 1;            // staging row 0..127
    const int skb  = (tid & 1) * 32;      // staging k sub-block {0,32}
    const bf16* aSrc = A  + (size_t)(m0 + srow) * lda + skb;
    const bf16* bSrc = Wt + (size_t)(n0 + srow) * ldw + skb;
    const unsigned ldsA[2] = {(unsigned)(size_t)&As[0][srow * LP + skb],
                              (unsigned)(size_t)&As[1][srow * LP + skb]};
    const unsigned ldsB[2] = {(unsigned)(size_t)&Bts[0][srow * LP + skb],
                              (unsigned)(size_t)&Bts[1][srow * LP + skb]};

    v8f acc[2][4];
    #pragma unroll
    for (int mi = 0; mi < 2; ++mi)
        #pragma unroll
        for (int ni = 0; ni < 4; ++ni) acc[mi][ni] = zero8();

    const int kbA = (lane >> 4) * 8;     // A-frag k sub-base within 32
    const int kbB = (lane >> 4) * 8;     // B-frag k sub-base within 32
    const int kTiles = K / BK;

    // prologue: DMA tile 0 into buffer 0
    async_copy64(ldsA[0], aSrc);
    async_copy64(ldsB[0], bSrc);

    for (int kt = 0; kt < kTiles; ++kt) {
        const int cur = kt & 1;
        wait_async0();        // this wave's DMA into buf[cur] complete
        __syncthreads();      // all waves' DMA complete, prev compute done

        if (kt + 1 < kTiles) {   // DMA next tile into the other buffer
            const int nxt = cur ^ 1;
            async_copy64(ldsA[nxt], aSrc + (size_t)(kt + 1) * BK);
            async_copy64(ldsB[nxt], bSrc + (size_t)(kt + 1) * BK);
        }

        // ---- compute on buf[cur]: 2 K-substeps x (2x4) WMMA tiles/wave ----
        #pragma unroll
        for (int ks = 0; ks < 2; ++ks) {
            const int kk = ks * 32;
            v16bf afr[2], bfr[4];
            #pragma unroll
            for (int mi = 0; mi < 2; ++mi) {
                const int row = wm + mi * 16 + (lane & 15);
                afr[mi] = cat8(*(const v8bf*)&As[cur][row * LP + kk + kbA],
                               *(const v8bf*)&As[cur][row * LP + kk + kbA + 16]);
            }
            #pragma unroll
            for (int ni = 0; ni < 4; ++ni) {
                const int col = wn + ni * 16 + (lane & 15);
                bfr[ni] = cat8(*(const v8bf*)&Bts[cur][col * LP + kk + kbB],
                               *(const v8bf*)&Bts[cur][col * LP + kk + kbB + 16]);
            }
            #pragma unroll
            for (int mi = 0; mi < 2; ++mi)
                #pragma unroll
                for (int ni = 0; ni < 4; ++ni)
                    acc[mi][ni] = wmma_bf16(afr[mi], bfr[ni], acc[mi][ni]);
        }
    }

    // ---- epilogue (no bounds checks: dims tile-aligned by padding) ----
    #pragma unroll
    for (int mi = 0; mi < 2; ++mi) {
        #pragma unroll
        for (int ni = 0; ni < 4; ++ni) {
            #pragma unroll
            for (int r = 0; r < 8; ++r) {
                const int row = m0 + wm + mi * 16 + r + 8 * (lane >> 4);
                const int col = n0 + wn + ni * 16 + (lane & 15);
                const float v = acc[mi][ni][r];
                if (mode == 0) {
                    Cbf[(size_t)row * ldc + col] = (bf16)v;
                } else {
                    Cf[(size_t)row * ldc + col] =
                        resid[(size_t)row * ldc + col] + v;
                }
            }
        }
    }
}

// ---------------------------------------------------------------------------
// Flash attention: causal softmax(QK^T/8) V, bf16 WMMA, fp32 online stats.
// grid (T/64, H, B), block 128 (4 waves); wave owns 16 query rows; the block
// shares a 32-key K/V step (V staged transposed in LDS).
// qkv layout: [B*T, 3D]; q @ col h*64, k @ D + h*64, v @ 2D + h*64.
// ---------------------------------------------------------------------------
#define ALP 40

__global__ __launch_bounds__(128) void attention_kernel(
    const bf16* __restrict__ qkv, const unsigned char* __restrict__ amask,
    bf16* __restrict__ out) {

    __shared__ bf16 Vt[HDIM * ALP];     // [hd][key] pitch 40
    __shared__ bf16 Ps[4][16 * ALP];    // per-wave P 16x32

    const int b    = blockIdx.z;
    const int h    = blockIdx.y;
    const int qblk = blockIdx.x;
    const int tid  = threadIdx.x;
    const int lane = tid & 31;
    const int w    = tid >> 5;
    const int q0   = qblk * 64 + w * 16;
    const int kbA  = (lane >> 4) * 8;
    const int kbB  = (lane >> 4) * 8;

    // Q fragments (16 rows x 64 hd = two K=32 A-fragments) in registers
    const int qrow = q0 + (lane & 15);
    const bf16* qp = qkv + ((size_t)(b * TT + qrow) * D3 + h * HDIM);
    const v16bf qa0 = cat8(ld8(qp + kbA),      ld8(qp + kbA + 16));
    const v16bf qa1 = cat8(ld8(qp + 32 + kbA), ld8(qp + 32 + kbA + 16));

    v8f o[4];
    #pragma unroll
    for (int i = 0; i < 4; ++i) o[i] = zero8();
    float mrow[8], lrow[8];
    #pragma unroll
    for (int r = 0; r < 8; ++r) { mrow[r] = -__builtin_inff(); lrow[r] = 0.f; }

    const int nkv = qblk * 2 + 2;   // causal: keys < q0_block + 64
    for (int kv = 0; kv < nkv; ++kv) {
        const int kvb = kv * 32;

        // ---- stage V block transposed: Vt[hd][key], 32 keys x 64 hd ----
        {
            const int key = tid >> 2;           // 0..31
            const int hd0 = (tid & 3) * 16;     // 0..48
            const bf16* vs = qkv +
                ((size_t)(b * TT + kvb + key) * D3 + 2 * DD + h * HDIM + hd0);
            v8bf v0 = ld8(vs);
            v8bf v1 = ld8(vs + 8);
            #pragma unroll
            for (int i = 0; i < 8; ++i) {
                Vt[(hd0 + i) * ALP + key]     = v0[i];
                Vt[(hd0 + 8 + i) * ALP + key] = v1[i];
            }
        }
        __syncthreads();

        // ---- S = Q @ K^T (16x32 per wave: 2 ntiles x 2 K-steps) ----
        v8f s[2];
        s[0] = zero8(); s[1] = zero8();
        #pragma unroll
        for (int nt = 0; nt < 2; ++nt) {
            const int key = kvb + nt * 16 + (lane & 15);
            const bf16* kp = qkv + ((size_t)(b * TT + key) * D3 + DD + h * HDIM);
            v16bf kf0 = cat8(ld8(kp + kbB),      ld8(kp + kbB + 16));
            v16bf kf1 = cat8(ld8(kp + 32 + kbB), ld8(kp + 32 + kbB + 16));
            s[nt] = wmma_bf16(qa0, kf0, s[nt]);
            s[nt] = wmma_bf16(qa1, kf1, s[nt]);
        }

        // ---- scale + causal + attention mask ----
        #pragma unroll
        for (int nt = 0; nt < 2; ++nt) {
            const int key = kvb + nt * 16 + (lane & 15);
            const bool km = (amask[b * TT + key] != 0);
            #pragma unroll
            for (int r = 0; r < 8; ++r) {
                const int row = q0 + r + 8 * (lane >> 4);
                const bool ok = km && (key <= row);
                s[nt][r] = ok ? s[nt][r] * 0.125f : -__builtin_inff();
            }
        }

        // ---- online softmax (each row lives in one 16-lane half) ----
        #pragma unroll
        for (int r = 0; r < 8; ++r) {
            float mx = fmaxf(s[0][r], s[1][r]);
            #pragma unroll
            for (int off = 8; off >= 1; off >>= 1)
                mx = fmaxf(mx, __shfl_xor(mx, off));
            const float mn = fmaxf(mrow[r], mx);
            float alpha, p0, p1;
            if (mn == -__builtin_inff()) {
                alpha = 1.f; p0 = 0.f; p1 = 0.f;
            } else {
                alpha = (mrow[r] == -__builtin_inff()) ? 0.f : expf(mrow[r] - mn);
                p0 = (s[0][r] == -__builtin_inff()) ? 0.f : expf(s[0][r] - mn);
                p1 = (s[1][r] == -__builtin_inff()) ? 0.f : expf(s[1][r] - mn);
            }
            float rs = p0 + p1;
            #pragma unroll
            for (int off = 8; off >= 1; off >>= 1) rs += __shfl_xor(rs, off);
            lrow[r] = lrow[r] * alpha + rs;
            mrow[r] = mn;
            #pragma unroll
            for (int nt = 0; nt < 4; ++nt) o[nt][r] *= alpha;
            const int prow = r + 8 * (lane >> 4);
            Ps[w][prow * ALP + (lane & 15)]      = (bf16)p0;
            Ps[w][prow * ALP + 16 + (lane & 15)] = (bf16)p1;
        }

        // ---- O += P @ V  (P 16x32, V 32x64 -> 4 ntiles) ----
        {
            const int prow = (lane & 15);
            v16bf pa = cat8(*(const v8bf*)&Ps[w][prow * ALP + kbA],
                            *(const v8bf*)&Ps[w][prow * ALP + kbA + 16]);
            #pragma unroll
            for (int nt = 0; nt < 4; ++nt) {
                const int hdc = nt * 16 + (lane & 15);
                v16bf vf = cat8(*(const v8bf*)&Vt[hdc * ALP + kbB],
                                *(const v8bf*)&Vt[hdc * ALP + kbB + 16]);
                o[nt] = wmma_bf16(pa, vf, o[nt]);
            }
        }
        __syncthreads();
    }

    // ---- epilogue: O /= l, write bf16 [B*T, D] at head offset ----
    #pragma unroll
    for (int r = 0; r < 8; ++r) {
        const float inv = (lrow[r] > 0.f) ? 1.f / lrow[r] : 0.f;
        const int row = q0 + r + 8 * (lane >> 4);
        #pragma unroll
        for (int nt = 0; nt < 4; ++nt) {
            out[(size_t)(b * TT + row) * DD + h * HDIM + nt * 16 + (lane & 15)] =
                (bf16)(o[nt][r] * inv);
        }
    }
}

// ---------------------------------------------------------------------------
// SwiGLU gate: g = silu(a) * c  (bf16, padded region is 0*0 = 0: safe)
// ---------------------------------------------------------------------------
__global__ void silu_mul_kernel(const bf16* __restrict__ a,
                                const bf16* __restrict__ c,
                                bf16* __restrict__ g, int n) {
    int i = blockIdx.x * blockDim.x + threadIdx.x;
    if (i < n) {
        float x = (float)a[i];
        float s = x / (1.f + expf(-x));
        g[i] = (bf16)(s * (float)c[i]);
    }
}

// ---------------------------------------------------------------------------
// Host-side orchestration
// ---------------------------------------------------------------------------
static inline int cdiv(int a, int b) { return (a + b - 1) / b; }

extern "C" void kernel_launch(void* const* d_in, const int* in_sizes, int n_in,
                              void* d_out, int out_size, void* d_ws,
                              size_t ws_size, hipStream_t stream) {
    (void)in_sizes; (void)n_in; (void)out_size; (void)ws_size;

    const float* x     = (const float*)d_in[0];
    const unsigned char* amask = (const unsigned char*)d_in[1];
    const float* Wqkv  = (const float*)d_in[2];
    const float* Wout  = (const float*)d_in[3];
    const float* W1    = (const float*)d_in[4];
    const float* W2    = (const float*)d_in[5];
    const float* W3    = (const float*)d_in[6];
    const float* g1    = (const float*)d_in[7];
    const float* b1    = (const float*)d_in[8];
    const float* g2    = (const float*)d_in[9];
    const float* b2    = (const float*)d_in[10];
    float* out         = (float*)d_out;

    // carve workspace (256B aligned)
    char* cur = (char*)d_ws;
    auto alloc = [&](size_t bytes) -> void* {
        void* p = cur;
        cur += (bytes + 255) & ~(size_t)255;
        return p;
    };
    bf16* WqkvT = (bf16*)alloc((size_t)D3 * DD * 2);     // [3072][1024]
    bf16* WoutT = (bf16*)alloc((size_t)DD * DD * 2);     // [1024][1024]
    bf16* W1T   = (bf16*)alloc((size_t)DFFP * DD * 2);   // [2816][1024]
    bf16* W2T   = (bf16*)alloc((size_t)DFFP * DD * 2);   // [2816][1024]
    bf16* W3T   = (bf16*)alloc((size_t)DD * DFFP * 2);   // [1024][2816]
    bf16* hbf   = (bf16*)alloc((size_t)BT * DD * 2);
    bf16* qkvbf = (bf16*)alloc((size_t)BT * D3 * 2);
    bf16* attnbf= (bf16*)alloc((size_t)BT * DD * 2);
    float* x1   = (float*)alloc((size_t)BT * DD * 4);
    bf16* t1bf  = (bf16*)alloc((size_t)BT * DFFP * 2);
    bf16* t2bf  = (bf16*)alloc((size_t)BT * DFFP * 2);

    // 1) weights -> transposed padded bf16
    transpose_pad_bf16_kernel<<<dim3(cdiv(DD, 32), cdiv(D3, 32)), 256, 0, stream>>>(
        Wqkv, DD, D3, WqkvT, DD);
    transpose_pad_bf16_kernel<<<dim3(cdiv(DD, 32), cdiv(DD, 32)), 256, 0, stream>>>(
        Wout, DD, DD, WoutT, DD);
    transpose_pad_bf16_kernel<<<dim3(cdiv(DD, 32), cdiv(DFFP, 32)), 256, 0, stream>>>(
        W1, DD, DFFN, W1T, DD);
    transpose_pad_bf16_kernel<<<dim3(cdiv(DD, 32), cdiv(DFFP, 32)), 256, 0, stream>>>(
        W2, DD, DFFN, W2T, DD);
    transpose_pad_bf16_kernel<<<dim3(cdiv(DFFP, 32), cdiv(DD, 32)), 256, 0, stream>>>(
        W3, DFFN, DD, W3T, DFFP);

    // 2) h = LN1(x)
    layernorm_kernel<<<BT, 256, 0, stream>>>(x, g1, b1, hbf);

    // 3) qkv = h @ W_qkv   [4096 x 3072]
    gemm_bf16_kernel<<<dim3(D3 / BN, BT / BM), 256, 0, stream>>>(
        hbf, DD, WqkvT, DD, DD, qkvbf, nullptr, nullptr, D3, 0);

    // 4) attention (flash, causal)
    attention_kernel<<<dim3(TT / 64, NH, 2), 128, 0, stream>>>(qkvbf, amask, attnbf);

    // 5) x1 = x + attn @ W_out
    gemm_bf16_kernel<<<dim3(DD / BN, BT / BM), 256, 0, stream>>>(
        attnbf, DD, WoutT, DD, DD, nullptr, x, x1, DD, 1);

    // 6) h2 = LN2(x1)   (reuse hbf)
    layernorm_kernel<<<BT, 256, 0, stream>>>(x1, g2, b2, hbf);

    // 7) t1 = h2 @ W1 ; t2 = h2 @ W2   [4096 x 2816(pad)]
    gemm_bf16_kernel<<<dim3(DFFP / BN, BT / BM), 256, 0, stream>>>(
        hbf, DD, W1T, DD, DD, t1bf, nullptr, nullptr, DFFP, 0);
    gemm_bf16_kernel<<<dim3(DFFP / BN, BT / BM), 256, 0, stream>>>(
        hbf, DD, W2T, DD, DD, t2bf, nullptr, nullptr, DFFP, 0);

    // 8) g = silu(t1) * t2  (in place on t1; pad region stays 0)
    {
        int n = BT * DFFP;
        silu_mul_kernel<<<cdiv(n, 256), 256, 0, stream>>>(t1bf, t2bf, t1bf, n);
    }

    // 9) out = x1 + g @ W3   (K = 2816 padded, zero rows inert)
    gemm_bf16_kernel<<<dim3(DD / BN, BT / BM), 256, 0, stream>>>(
        t1bf, DFFP, W3T, DFFP, DFFP, nullptr, x1, out, DD, 1);
}
